// GRUDEncoder_45037027065873
// MI455X (gfx1250) — compile-verified, matrix operand order checked by reading, not run on hardware
//
#include <hip/hip_runtime.h>
#include <hip/hip_bf16.h>

// ---------------------------------------------------------------------------
// GRU-D encoder for MI455X (gfx1250, wave32, WMMA + async global->LDS copies)
// B=256, S=200, D=64, H=512, TD=8
// ---------------------------------------------------------------------------

typedef __bf16 bf16;
typedef __attribute__((ext_vector_type(16))) __bf16 v16bf;
typedef __attribute__((ext_vector_type(8)))  float  v8f;

constexpr int cB  = 256;
constexpr int cS  = 200;
constexpr int cD  = 64;
constexpr int cH  = 512;
constexpr int cTD = 8;
constexpr int cH3 = 3 * cH;          // 1536
constexpr int cKO = 544;             // 520 (H+TD) padded up to multiple of 32

#if defined(__HIP_DEVICE_COMPILE__) && defined(__gfx1250__)
#define USE_ASYNC_COPY 1
#else
#define USE_ASYNC_COPY 0
#endif

// 32-byte global -> LDS copy.  On gfx1250 use the direct async-to-LDS path
// (no VGPR round trip, tracked by ASYNCcnt).  The 24-bit immediate offset is
// added to BOTH the LDS and the memory address (ISA 15.18.3), so offset:16
// covers the second 16-byte half.
__device__ __forceinline__ void copy32_to_lds(const bf16* __restrict__ g,
                                              bf16* l) {
#if USE_ASYNC_COPY
  unsigned           ldsoff = (unsigned)(unsigned long long)(uintptr_t)l;
  unsigned long long gaddr  = (unsigned long long)(uintptr_t)g;
  asm volatile("global_load_async_to_lds_b128 %0, %1, off\n\t"
               "global_load_async_to_lds_b128 %0, %1, off offset:16"
               :
               : "v"(ldsoff), "v"(gaddr)
               : "memory");
#else
  *(v16bf*)l = *(const v16bf*)g;
#endif
}

__device__ __forceinline__ void async_copies_wait() {
#if USE_ASYNC_COPY
  asm volatile("s_wait_asynccnt 0x0" ::: "memory");
#endif
}

// ---------------------------------------------------------------------------
// small utility kernels
// ---------------------------------------------------------------------------

__global__ void f32_to_bf16_kernel(const float* __restrict__ src,
                                   bf16* __restrict__ dst, int n) {
  int i = blockIdx.x * 256 + threadIdx.x;
  if (i < n) dst[i] = (bf16)src[i];
}

// pack W_out (H x (H+TD)) -> bf16 (H x cKO) with zero padding in K
__global__ void pack_wout_kernel(const float* __restrict__ w,
                                 bf16* __restrict__ dst) {
  int i = blockIdx.x * 256 + threadIdx.x;
  if (i < cH * cKO) {
    int r = i / cKO, c = i % cKO;
    dst[i] = (bf16)((c < cH + cTD) ? w[r * (cH + cTD) + c] : 0.f);
  }
}

// x_mean[d] = sum_{b,s} mask*x / max(sum mask, 1)
__global__ void xmean_kernel(const float* __restrict__ x,
                             const float* __restrict__ mask,
                             float* __restrict__ xmean) {
  const int d = blockIdx.x;          // 0..63
  const int tid = threadIdx.x;       // 256 threads
  float sm = 0.f, sw = 0.f;
  for (int i = tid; i < cB * cS; i += 256) {
    float m = mask[(size_t)i * cD + d];
    sm += m * x[(size_t)i * cD + d];
    sw += m;
  }
  __shared__ float s1[256], s2[256];
  s1[tid] = sm; s2[tid] = sw;
  __syncthreads();
  for (int k = 128; k > 0; k >>= 1) {
    if (tid < k) { s1[tid] += s1[tid + k]; s2[tid] += s2[tid + k]; }
    __syncthreads();
  }
  if (tid == 0) xmean[d] = s1[0] / fmaxf(s2[0], 1.f);
}

// per-batch sequential scans: delta, x_last, x_hat -> time-major bf16 buffers
__global__ void preproc_kernel(const float* __restrict__ x,
                               const float* __restrict__ tarr,
                               const float* __restrict__ mask,
                               const float* __restrict__ w_dg_x,
                               const float* __restrict__ b_dg_x,
                               const float* __restrict__ xmean,
                               bf16* __restrict__ xhat_b,
                               bf16* __restrict__ delta_b,
                               bf16* __restrict__ m_b) {
  const int b = blockIdx.x;          // batch
  const int d = threadIdx.x;         // 0..63 (2 waves)
  const float xm  = xmean[d];
  const float wdg = w_dg_x[d];
  const float bdg = b_dg_x[d];
  float x_last = 0.f, dlt = 0.f, mprev = 1.f;
  float tprev = tarr[(size_t)b * cS];     // dt[0] = 0 (prepend t[:, :1])
  for (int t = 0; t < cS; ++t) {
    float tc = tarr[(size_t)b * cS + t];
    float dt = tc - tprev; tprev = tc;
    float m  = mask[((size_t)b * cS + t) * cD + d];
    float xv = x   [((size_t)b * cS + t) * cD + d];
    dlt = dt + (1.f - mprev) * dlt;
    mprev = m;
    float gx = __expf(-fmaxf(wdg * dlt + bdg, 0.f));
    float xh = m * xv + (1.f - m) * (gx * x_last + (1.f - gx) * xm);
    x_last   = m * xv + (1.f - m) * x_last;
    size_t o = ((size_t)t * cB + b) * cD + d;   // time-major
    xhat_b[o]  = (bf16)xh;
    delta_b[o] = (bf16)dlt;
    m_b[o]     = (bf16)m;
  }
}

// ---------------------------------------------------------------------------
// generic WMMA bf16 GEMM:  out[M,N] = sum_p A_p[M,K] * W_p[N,K]^T (+bias)(+act)
// workgroup tile 128x64, 8 waves, each wave 32x32 (4 x v_wmma 16x16x32 bf16)
// M % 128 == 0, N % 64 == 0, K % 32 == 0 (all shapes here satisfy this)
// EPI: 0 -> fp32 out, 1 -> bf16 out, 2 -> bf16 exp(-relu(.)) out
// ---------------------------------------------------------------------------
template <int NPAIR, int EPI>
__global__ __launch_bounds__(256) void wmma_gemm_kernel(
    const bf16* __restrict__ A0, const bf16* __restrict__ W0,
    const bf16* __restrict__ A1, const bf16* __restrict__ W1,
    const float* __restrict__ bias, void* __restrict__ outv,
    int M, int N, int K) {
  __shared__ bf16 Alds[128 * 32];
  __shared__ bf16 Blds[64 * 32];
  const int tid  = threadIdx.x;
  const int lane = tid & 31;
  const int wave = tid >> 5;       // 0..7
  const int wm   = wave >> 1;      // 0..3  (M direction, 32 rows each)
  const int wn   = wave & 1;       // 0..1  (N direction, 32 cols each)
  const int m0   = blockIdx.x * 128;
  const int n0   = blockIdx.y * 64;
  const int ra   = tid >> 1;            // A stage row for this thread
  const int ca   = (tid & 1) * 16;      // A stage col

  v8f acc[2][2] = {};

#pragma unroll
  for (int p = 0; p < NPAIR; ++p) {
    const bf16* A  = p ? A1 : A0;
    const bf16* Wt = p ? W1 : W0;
    for (int k0 = 0; k0 < K; k0 += 32) {
      // stage A tile 128x32 and B tile 64x32 directly into LDS (async path)
      copy32_to_lds(&A[(size_t)(m0 + ra) * K + k0 + ca], &Alds[ra * 32 + ca]);
      if (tid < 128)
        copy32_to_lds(&Wt[(size_t)(n0 + ra) * K + k0 + ca],
                      &Blds[ra * 32 + ca]);
      // pull next K-tile of A toward the caches while WMMA runs
      if (k0 + 32 < K)
        __builtin_prefetch(&A[(size_t)(m0 + ra) * K + k0 + 32 + ca], 0, 1);
      async_copies_wait();
      __syncthreads();

      v16bf af[2], bfm[2];
#pragma unroll
      for (int mi = 0; mi < 2; ++mi) {
        int row = wm * 32 + mi * 16 + (lane & 15);
        af[mi] = *(const v16bf*)&Alds[row * 32 + (lane >> 4) * 16];
      }
#pragma unroll
      for (int ni = 0; ni < 2; ++ni) {
        int nn = wn * 32 + ni * 16 + (lane & 15);
        bfm[ni] = *(const v16bf*)&Blds[nn * 32 + (lane >> 4) * 16];
      }
#pragma unroll
      for (int mi = 0; mi < 2; ++mi)
#pragma unroll
        for (int ni = 0; ni < 2; ++ni)
          acc[mi][ni] = __builtin_amdgcn_wmma_f32_16x16x32_bf16(
              false, af[mi], false, bfm[ni], (short)0, acc[mi][ni],
              false, false);
      __syncthreads();
    }
  }

  // epilogue: ISA C-layout — VGPR i, lanes 0-15: M=i, N=lane; 16-31: M=8+i
#pragma unroll
  for (int mi = 0; mi < 2; ++mi) {
#pragma unroll
    for (int ni = 0; ni < 2; ++ni) {
#pragma unroll
      for (int i = 0; i < 8; ++i) {
        int row = m0 + wm * 32 + mi * 16 + (lane >> 4) * 8 + i;
        int col = n0 + wn * 32 + ni * 16 + (lane & 15);
        float v = acc[mi][ni][i];
        if (bias) v += bias[col];
        if (EPI == 2) v = __expf(-fmaxf(v, 0.f));
        size_t o = (size_t)row * N + col;
        if (EPI == 0) ((float*)outv)[o] = v;
        else          ((bf16*)outv)[o] = (bf16)v;
      }
    }
  }
}

// ---------------------------------------------------------------------------
// recurrent gate update (fused with pre-scaling of h_dec for the next step)
// ---------------------------------------------------------------------------
__global__ void init_hdec_kernel(float* __restrict__ hf, bf16* __restrict__ hb) {
  int i = blockIdx.x * 256 + threadIdx.x;
  hf[i] = 0.f;
  hb[i] = (bf16)0.f;
}

__global__ void gru_update_kernel(const float* __restrict__ ACC,
                                  const bf16* __restrict__ PRE,
                                  const bf16* __restrict__ gh_next,
                                  float* __restrict__ hdec_f,
                                  bf16* __restrict__ hdec_b,
                                  bf16* __restrict__ h_out,
                                  int has_next) {
  int i = blockIdx.x * 256 + threadIdx.x;   // 0..131071
  int b = i >> 9, j = i & (cH - 1);
  size_t base = (size_t)b * cH3;
  float z = ACC[base + j] + (float)PRE[base + j];
  z = 1.f / (1.f + __expf(-z));
  float r = ACC[base + cH + j] + (float)PRE[base + cH + j];
  r = 1.f / (1.f + __expf(-r));
  float ht = tanhf((float)PRE[base + 2 * cH + j] + r * ACC[base + 2 * cH + j]);
  float hd = hdec_f[i];
  float hn = (1.f - z) * hd + z * ht;
  h_out[i] = (bf16)hn;
  if (has_next) {
    float hdn = (float)gh_next[i] * hn;
    hdec_f[i] = hdn;
    hdec_b[i] = (bf16)hdn;
  }
}

// ---------------------------------------------------------------------------
// attention over time + build padded concat buffer (bf16, K = cKO)
// ---------------------------------------------------------------------------
__global__ void attn_kernel(const bf16* __restrict__ h_all,
                            const float* __restrict__ attn_v,
                            const float* __restrict__ a,
                            bf16* __restrict__ concat) {
  const int b = blockIdx.x, tid = threadIdx.x;
  __shared__ float sc[cS];
  __shared__ float inv_s;
  for (int t = tid; t < cS; t += 256) {
    const bf16* hp = h_all + ((size_t)t * cB + b) * cH;
    float acc = 0.f;
    for (int j = 0; j < cH; ++j) acc += (float)hp[j] * attn_v[j];
    sc[t] = acc * 0.04419417382415922f;   // 1/sqrt(512)
  }
  __syncthreads();
  if (tid == 0) {
    float mx = -1e30f;
    for (int t = 0; t < cS; ++t) mx = fmaxf(mx, sc[t]);
    float sum = 0.f;
    for (int t = 0; t < cS; ++t) { float e = __expf(sc[t] - mx); sc[t] = e; sum += e; }
    inv_s = 1.f / sum;
  }
  __syncthreads();
  float inv = inv_s;
  for (int j = tid; j < cH; j += 256) {
    float acc = 0.f;
    for (int t = 0; t < cS; ++t)
      acc += sc[t] * (float)h_all[((size_t)t * cB + b) * cH + j];
    concat[(size_t)b * cKO + j] = (bf16)(acc * inv);
  }
  if (tid < cTD)
    concat[(size_t)b * cKO + cH + tid] = (bf16)a[(size_t)b * cTD + tid];
  else if (tid < cKO - cH)
    concat[(size_t)b * cKO + cH + tid] = (bf16)0.f;
}

// ---------------------------------------------------------------------------
// host launcher
// ---------------------------------------------------------------------------
extern "C" void kernel_launch(void* const* d_in, const int* in_sizes, int n_in,
                              void* d_out, int out_size, void* d_ws,
                              size_t ws_size, hipStream_t stream) {
  (void)in_sizes; (void)n_in; (void)out_size; (void)ws_size;
  const float* x      = (const float*)d_in[0];
  const float* a      = (const float*)d_in[1];
  const float* tarr   = (const float*)d_in[2];
  const float* mask   = (const float*)d_in[3];
  const float* w_dg_x = (const float*)d_in[4];
  const float* b_dg_x = (const float*)d_in[5];
  const float* W_dg_h = (const float*)d_in[6];
  const float* b_dg_h = (const float*)d_in[7];
  const float* W      = (const float*)d_in[8];
  const float* U      = (const float*)d_in[9];
  const float* V      = (const float*)d_in[10];
  const float* bgate  = (const float*)d_in[11];
  const float* attn_v = (const float*)d_in[12];
  const float* W_out  = (const float*)d_in[13];
  const float* b_out  = (const float*)d_in[14];

  // ---- carve workspace -------------------------------------------------
  char* p = (char*)d_ws;
  auto alloc = [&](size_t bytes) -> char* {
    char* r = p;
    p += (bytes + 255) & ~(size_t)255;
    return r;
  };
  const size_t MSB = (size_t)cS * cB;          // 51200 rows

  bf16*  W_bf    = (bf16*)alloc((size_t)cH3 * cD * 2);
  bf16*  V_bf    = (bf16*)alloc((size_t)cH3 * cD * 2);
  bf16*  U_bf    = (bf16*)alloc((size_t)cH3 * cH * 2);
  bf16*  Wdg_bf  = (bf16*)alloc((size_t)cH * cD * 2);
  bf16*  Wout_bf = (bf16*)alloc((size_t)cH * cKO * 2);
  float* xmean   = (float*)alloc(cD * 4);
  bf16*  xhat_b  = (bf16*)alloc(MSB * cD * 2);
  bf16*  delta_b = (bf16*)alloc(MSB * cD * 2);
  bf16*  m_b     = (bf16*)alloc(MSB * cD * 2);
  bf16*  PRE     = (bf16*)alloc(MSB * cH3 * 2);   // x-side gate pre-acts
  bf16*  GH      = (bf16*)alloc(MSB * cH * 2);    // gamma_h
  bf16*  h_all   = (bf16*)alloc(MSB * cH * 2);
  float* ACC     = (float*)alloc((size_t)cB * cH3 * 4);
  float* hdec_f  = (float*)alloc((size_t)cB * cH * 4);
  bf16*  hdec_b  = (bf16*)alloc((size_t)cB * cH * 2);
  bf16*  concat  = (bf16*)alloc((size_t)cB * cKO * 2);

  // ---- weight packing --------------------------------------------------
  auto cvt = [&](const float* s, bf16* d, int n) {
    f32_to_bf16_kernel<<<(n + 255) / 256, 256, 0, stream>>>(s, d, n);
  };
  cvt(W, W_bf, cH3 * cD);
  cvt(V, V_bf, cH3 * cD);
  cvt(U, U_bf, cH3 * cH);
  cvt(W_dg_h, Wdg_bf, cH * cD);
  pack_wout_kernel<<<(cH * cKO + 255) / 256, 256, 0, stream>>>(W_out, Wout_bf);

  // ---- prologue --------------------------------------------------------
  xmean_kernel<<<cD, 256, 0, stream>>>(x, mask, xmean);
  preproc_kernel<<<cB, cD, 0, stream>>>(x, tarr, mask, w_dg_x, b_dg_x, xmean,
                                        xhat_b, delta_b, m_b);

  // PRE = x_hat@W^T + m@V^T + b      (M=51200, N=1536, K=64), bf16 out
  wmma_gemm_kernel<2, 1><<<dim3(MSB / 128, cH3 / 64), 256, 0, stream>>>(
      xhat_b, W_bf, m_b, V_bf, bgate, PRE, (int)MSB, cH3, cD);
  // GH = exp(-relu(delta@W_dg_h^T + b_dg_h))   (M=51200, N=512, K=64)
  wmma_gemm_kernel<1, 2><<<dim3(MSB / 128, cH / 64), 256, 0, stream>>>(
      delta_b, Wdg_bf, nullptr, nullptr, b_dg_h, GH, (int)MSB, cH, cD);

  // ---- recurrence ------------------------------------------------------
  init_hdec_kernel<<<(cB * cH) / 256, 256, 0, stream>>>(hdec_f, hdec_b);
  for (int t = 0; t < cS; ++t) {
    // ACC = h_dec @ U^T  (M=256, N=1536, K=512), fp32 out
    wmma_gemm_kernel<1, 0><<<dim3(cB / 128, cH3 / 64), 256, 0, stream>>>(
        hdec_b, U_bf, nullptr, nullptr, nullptr, ACC, cB, cH3, cH);
    const bf16* gh_next = (t + 1 < cS) ? (GH + (size_t)(t + 1) * cB * cH) : GH;
    gru_update_kernel<<<(cB * cH) / 256, 256, 0, stream>>>(
        ACC, PRE + (size_t)t * cB * cH3, gh_next, hdec_f, hdec_b,
        h_all + (size_t)t * cB * cH, (t + 1 < cS) ? 1 : 0);
  }

  // ---- attention + output ---------------------------------------------
  attn_kernel<<<cB, 256, 0, stream>>>(h_all, attn_v, a, concat);
  // out = concat @ W_out^T + b_out  (M=256, N=512, K=544), fp32 -> d_out
  wmma_gemm_kernel<1, 0><<<dim3(cB / 128, cH / 64), 256, 0, stream>>>(
      concat, Wout_bf, nullptr, nullptr, b_out, (float*)d_out, cB, cH, cKO);
}
